// MiniAttention_52802327937636
// MI455X (gfx1250) — matrix operand order repaired and checked by reading, not compiled, not run on hardware
//
#include <hip/hip_runtime.h>
#include <hip/hip_bf16.h>
#include <stdint.h>

// ---------------------------------------------------------------------------
// MiniAttention for MI455X (gfx1250).
//  * all matrix math on v_wmma_f32_16x16x32_bf16 (fp32 accumulate)
//  * per-block A-strip staged into LDS by the Tensor Data Mover
//    (tensor_load_to_lds + s_wait_tensorcnt), B streamed from global with a
//    software-pipelined (double-buffered) K loop + global_prefetch_b8
//  * fused 12x12 head-mix convs + softmax in LDS
// Workload is HBM-bound (~0.5 GB traffic vs ~36 GFLOP), hence bf16 operands
// and layouts that make every fragment load a contiguous 16B access.
// ---------------------------------------------------------------------------

#define DEVINL __device__ __forceinline__

typedef __attribute__((ext_vector_type(16))) __bf16 v16bf;
typedef __attribute__((ext_vector_type(8)))  __bf16 v8bf;
typedef __attribute__((ext_vector_type(8)))  float  v8f;
typedef __attribute__((ext_vector_type(4)))  unsigned int v4u;
typedef __attribute__((ext_vector_type(8)))  int    v8i;
typedef __attribute__((ext_vector_type(4)))  int    v4i;

constexpr int BB = 32;          // batch
constexpr int NN = 197;         // tokens
constexpr int CC = 768;         // channels
constexpr int HH = 12;          // heads
constexpr int DD = 64;          // head dim
constexpr int NP = 208;         // N padded to 16  (13 tiles)
constexpr int NK = 224;         // N padded to 32  (K dim of attn_w @ v)
constexpr int BH = BB * HH;     // 384
constexpr int C3 = 3 * CC;      // 2304
constexpr float SCALE = 0.125f; // 64^-0.5

// fp32 -> bf16 round-to-nearest-even
DEVINL __bf16 f2bf(float f) {
  unsigned u = __builtin_bit_cast(unsigned, f);
  unsigned r = u + 0x7FFFu + ((u >> 16) & 1u);
  unsigned short h = (unsigned short)(r >> 16);
  return __builtin_bit_cast(__bf16, h);
}

// ---- WMMA fragment loaders (layouts per CDNA5 ISA §7.12.2) -----
// A 16x32 bf16: lane<16 = row M=lane, K {0..7,16..23}; lane>=16 = row with
// K {8..15,24..31}.  rp points at this lane's row (global or LDS).
DEVINL v16bf load_frag_a(const __bf16* rp, int kc, int half) {
  union { v16bf v; v8bf h[2]; } t;
  t.h[0] = *reinterpret_cast<const v8bf*>(rp + kc + half * 8);
  t.h[1] = *reinterpret_cast<const v8bf*>(rp + kc + 16 + half * 8);
  return t.v;
}
// B 32x16 bf16: lane<16 = column, K {0..7,16..23}; lane>=16 = column,
// K {8..15,24..31} (same order as A).  Source stored row-major [col][k].
DEVINL v16bf load_frag_b(const __bf16* rp, int kc, int half) {
  union { v16bf v; v8bf h[2]; } t;
  t.h[0] = *reinterpret_cast<const v8bf*>(rp + kc + half * 8);
  t.h[1] = *reinterpret_cast<const v8bf*>(rp + kc + 16 + half * 8);
  return t.v;
}

// 16x16 fp32 tile over K (multiple of 32). A typically from LDS (ds_load),
// B from global; B loads double-buffered so each WMMA only waits on the
// previous chunk's loads, not the ones just issued.
template <int KT>
DEVINL v8f wmma_tile(const __bf16* aRow, const __bf16* __restrict__ bRow,
                     int half) {
  v8f c = {0.f, 0.f, 0.f, 0.f, 0.f, 0.f, 0.f, 0.f};
  v16bf b0 = load_frag_b(bRow, 0, half);
#pragma unroll
  for (int kc = 0; kc < KT; kc += 32) {
    v16bf a = load_frag_a(aRow, kc, half);
    v16bf b1 = b0;
    if (kc + 32 < KT) {
      b1 = load_frag_b(bRow, kc + 32, half);
      if (kc + 256 < KT)                       // compile-time: no branches
        __builtin_prefetch(bRow + kc + 256, 0, 3);
    }
    c = __builtin_amdgcn_wmma_f32_16x16x32_bf16(false, a, false, b0, (short)0,
                                                c, false, false);
    b0 = b1;
  }
  return c;
}

// ---- Tensor Data Mover: DMA a [rows x rowElems] bf16 strip into LDS -------
// D# per CDNA5 ISA (2D tensor == tile, stride = row).
DEVINL void tdm_load_2d(uint32_t lds_off, const __bf16* gp, uint32_t rows,
                        uint32_t rowElems) {
  uint64_t ga = (uint64_t)(uintptr_t)gp;
  v4u g0;
  g0[0] = 1u;                                            // count=1 (user D#)
  g0[1] = lds_off;                                       // lds_addr
  g0[2] = (uint32_t)ga;                                  // global_addr[31:0]
  g0[3] = (uint32_t)((ga >> 32) & 0x01FFFFFFu) | (2u << 30);  // [56:32]|type=2
  v8i g1;
  g1[0] = (int)(1u << 16);                               // data_size = 2B
  g1[1] = (int)((rowElems & 0xFFFFu) << 16);             // tensor_dim0 lo
  g1[2] = (int)(((rowElems >> 16) & 0xFFFFu) | ((rows & 0xFFFFu) << 16));
  g1[3] = (int)(((rows >> 16) & 0xFFFFu) | ((rowElems & 0xFFFFu) << 16));
  g1[4] = (int)(rows & 0xFFFFu);                         // tile_dim1
  g1[5] = (int)rowElems;                                 // dim0_stride lo32
  g1[6] = 0;
  g1[7] = 0;
  v4i z4 = {0, 0, 0, 0};
#if defined(__clang_major__) && __clang_major__ >= 23
  v8i z8 = {0, 0, 0, 0, 0, 0, 0, 0};
  __builtin_amdgcn_tensor_load_to_lds(g0, g1, z4, z4, z8, 0);
#else
  __builtin_amdgcn_tensor_load_to_lds(g0, g1, z4, z4, 0);
#endif
}

// ---------------------------- prep kernels ---------------------------------
__global__ void __launch_bounds__(256) prep_x_kernel(
    const float* __restrict__ x, __bf16* __restrict__ xb) {
  size_t i = (size_t)blockIdx.x * 256 + threadIdx.x;
  if (i >= (size_t)BB * NP * CC) return;
  int c = (int)(i % CC);
  size_t t = i / CC;
  int n = (int)(t % NP);
  int b = (int)(t / NP);
  float v = (n < NN) ? x[((size_t)b * NN + n) * CC + c] : 0.f;
  xb[i] = f2bf(v);
}

__global__ void __launch_bounds__(256) prep_wT_kernel(
    const float* __restrict__ w, __bf16* __restrict__ wt, int K, int Nc) {
  size_t i = (size_t)blockIdx.x * 256 + threadIdx.x;
  if (i >= (size_t)K * Nc) return;
  int k = (int)(i % K);
  int j = (int)(i / K);
  wt[i] = f2bf(w[(size_t)k * Nc + j]);
}

__global__ void __launch_bounds__(256) zero_u32_kernel(
    uint32_t* __restrict__ p, size_t n) {
  for (size_t i = (size_t)blockIdx.x * 256 + threadIdx.x; i < n;
       i += (size_t)gridDim.x * 256)
    p[i] = 0u;
}

// ------------------- QKV projection: x @ w_qkv -----------------------------
__global__ void __launch_bounds__(256) gemm_qkv_kernel(
    const __bf16* __restrict__ xb, const __bf16* __restrict__ wqkvT,
    __bf16* __restrict__ qb, __bf16* __restrict__ kb,
    __bf16* __restrict__ vb, __bf16* __restrict__ vt) {
  __shared__ __bf16 ldsA[16 * CC];                 // 24 KB A strip
  const int rt = blockIdx.x / 18;                  // b*13 + nt
  const int cg = blockIdx.x % 18;
  const int wave = threadIdx.x >> 5, lane = threadIdx.x & 31;
  const int half = lane >> 4, r = lane & 15;
  const int b = rt / 13, nt = rt % 13;
  if (wave == 0) {                                 // one TDM per block
    tdm_load_2d((uint32_t)(uintptr_t)ldsA,
                xb + (size_t)(b * NP + nt * 16) * CC, 16, CC);
    __builtin_amdgcn_s_wait_tensorcnt(0);
  }
  __syncthreads();
  const int colTile = cg * 8 + wave;               // 0..143
  const __bf16* aRow = ldsA + r * CC;
  const __bf16* bRow = wqkvT + (size_t)(colTile * 16 + r) * CC;
  v8f acc = wmma_tile<CC>(aRow, bRow, half);
  const int j = colTile * 16 + r;
  const int which = j / CC;
  const int rem = j % CC;
  const int h = rem >> 6, d = rem & 63;
  const size_t bh = (size_t)b * HH + h;
#pragma unroll
  for (int i = 0; i < 8; ++i) {
    const int n = nt * 16 + i + 8 * half;          // < 208
    const float val = (n < NN) ? acc[i] : 0.f;
    const __bf16 bv = f2bf(val);
    const size_t qi = (bh * NP + n) * DD + d;
    if (which == 0)      qb[qi] = bv;
    else if (which == 1) kb[qi] = bv;
    else { vb[qi] = bv; vt[(bh * DD + d) * NK + n] = bv; }
  }
}

// -------- batched X @ Y^T (K=64): out = alpha * A·B^T, fp32 ----------------
__global__ void __launch_bounds__(256) gemm_nt_kernel(
    const __bf16* __restrict__ Ab, const __bf16* __restrict__ Bb,
    float* __restrict__ outp, float alpha) {
  __shared__ __bf16 ldsA[16 * DD];                 // 2 KB A strip
  const int bh = blockIdx.x / 26;
  const int rm = blockIdx.x % 26;
  const int nt = rm >> 1, cg = rm & 1;
  const int wave = threadIdx.x >> 5, lane = threadIdx.x & 31;
  const int half = lane >> 4, r = lane & 15;
  if (wave == 0) {
    tdm_load_2d((uint32_t)(uintptr_t)ldsA,
                Ab + ((size_t)bh * NP + nt * 16) * DD, 16, DD);
    __builtin_amdgcn_s_wait_tensorcnt(0);
  }
  __syncthreads();
  const int ct = cg * 8 + wave;
  const int ctc = (ct < 13) ? ct : 12;             // clamp; stores guarded
  const __bf16* aRow = ldsA + r * DD;
  const __bf16* bRow = Bb + ((size_t)bh * NP + ctc * 16 + r) * DD;
  v8f acc = wmma_tile<DD>(aRow, bRow, half);
  const int m = ct * 16 + r;
  if (ct < 13 && m < NN) {
#pragma unroll
    for (int i = 0; i < 8; ++i) {
      const int n = nt * 16 + i + 8 * half;
      if (n < NN) outp[((size_t)bh * NN + n) * NN + m] = alpha * acc[i];
    }
  }
}

// ------------- in-place row softmax over last dim (len 197) ----------------
__global__ void __launch_bounds__(256) softmax_rows_kernel(
    float* __restrict__ data, int nrows) {
  const int gw = (int)((blockIdx.x * 256 + threadIdx.x) >> 5);
  const int lane = threadIdx.x & 31;
  if (gw >= nrows) return;
  float* row = data + (size_t)gw * NN;
  float mx = -__builtin_inff();
  for (int m = lane; m < NN; m += 32) mx = fmaxf(mx, row[m]);
#pragma unroll
  for (int off = 16; off > 0; off >>= 1) mx = fmaxf(mx, __shfl_xor(mx, off, 32));
  float e[7];
  int cnt = 0;
  float sum = 0.f;
  for (int m = lane; m < NN; m += 32) {
    float v = __expf(row[m] - mx);
    e[cnt++] = v;
    sum += v;
  }
#pragma unroll
  for (int off = 16; off > 0; off >>= 1) sum += __shfl_xor(sum, off, 32);
  const float inv = 1.f / sum;
  cnt = 0;
  for (int m = lane; m < NN; m += 32) row[m] = e[cnt++] * inv;
}

// ---- fused: conv_l (12x12 head mix) -> softmax -> probs out -> conv_w -----
__global__ void __launch_bounds__(256) conv_softmax_kernel(
    const float* __restrict__ scores, const float* __restrict__ wl,
    const float* __restrict__ ww, float* __restrict__ probs,
    __bf16* __restrict__ attnw) {
  __shared__ float s[HH][NP];
  __shared__ float p[HH][NP];
  __shared__ float wls[HH * HH];
  __shared__ float wws[HH * HH];
  const int b = blockIdx.x / NN, n = blockIdx.x % NN;
  const int tid = threadIdx.x;
  if (tid < HH * HH) { wls[tid] = wl[tid]; wws[tid] = ww[tid]; }
  for (int idx = tid; idx < HH * NN; idx += 256) {
    int h = idx / NN, m = idx % NN;
    s[h][m] = scores[(((size_t)b * HH + h) * NN + n) * NN + m];
  }
  __syncthreads();
  if (tid < NN) {
    float sv[HH];
#pragma unroll
    for (int h = 0; h < HH; ++h) sv[h] = s[h][tid];
#pragma unroll
    for (int o = 0; o < HH; ++o) {
      float acc = 0.f;
#pragma unroll
      for (int h = 0; h < HH; ++h) acc += wls[o * HH + h] * sv[h];
      p[o][tid] = acc;
    }
  }
  __syncthreads();
  const int wave = tid >> 5, lane = tid & 31;
  for (int o = wave; o < HH; o += 8) {
    float mx = -__builtin_inff();
    for (int m = lane; m < NN; m += 32) mx = fmaxf(mx, p[o][m]);
#pragma unroll
    for (int off = 16; off > 0; off >>= 1) mx = fmaxf(mx, __shfl_xor(mx, off, 32));
    float sum = 0.f;
    for (int m = lane; m < NN; m += 32) {
      float e = __expf(p[o][m] - mx);
      p[o][m] = e;
      sum += e;
    }
#pragma unroll
    for (int off = 16; off > 0; off >>= 1) sum += __shfl_xor(sum, off, 32);
    const float inv = 1.f / sum;
    for (int m = lane; m < NN; m += 32) {
      float pr = p[o][m] * inv;
      p[o][m] = pr;
      probs[(((size_t)b * HH + o) * NN + n) * NN + m] = pr;
    }
  }
  __syncthreads();
  if (tid < NK) {
    if (tid < NN) {
      float pv[HH];
#pragma unroll
      for (int h = 0; h < HH; ++h) pv[h] = p[h][tid];
#pragma unroll
      for (int o = 0; o < HH; ++o) {
        float acc = 0.f;
#pragma unroll
        for (int h = 0; h < HH; ++h) acc += wws[o * HH + h] * pv[h];
        attnw[(((size_t)b * HH + o) * NP + n) * NK + tid] = f2bf(acc);
      }
    } else {
#pragma unroll
      for (int o = 0; o < HH; ++o)
        attnw[(((size_t)b * HH + o) * NP + n) * NK + tid] = f2bf(0.f);
    }
  }
}

// ----------------- out = attn_w @ v  (K = NK = 224) ------------------------
__global__ void __launch_bounds__(256) gemm_av_kernel(
    const __bf16* __restrict__ aw, const __bf16* __restrict__ vt,
    __bf16* __restrict__ outb) {
  __shared__ __bf16 ldsA[2 * 16 * NK];             // two 7 KB A strips
  const int bh = blockIdx.x / 7;
  const int rg = blockIdx.x % 7;
  const int wave = threadIdx.x >> 5, lane = threadIdx.x & 31;
  const int half = lane >> 4, r = lane & 15;
  const int rt = rg * 2 + (wave >> 2);
  if (wave == 0) {
    tdm_load_2d((uint32_t)(uintptr_t)ldsA,
                aw + ((size_t)bh * NP + (rg * 2) * 16) * NK, 16, NK);
    __builtin_amdgcn_s_wait_tensorcnt(0);
  } else if (wave == 4 && rt < 13) {
    tdm_load_2d((uint32_t)(uintptr_t)(ldsA + 16 * NK),
                aw + ((size_t)bh * NP + (rg * 2 + 1) * 16) * NK, 16, NK);
    __builtin_amdgcn_s_wait_tensorcnt(0);
  }
  __syncthreads();
  const int ct = wave & 3;
  const __bf16* aRow = ldsA + (wave >> 2) * (16 * NK) + r * NK;
  const __bf16* bRow = vt + ((size_t)bh * DD + ct * 16 + r) * NK;
  v8f acc = wmma_tile<NK>(aRow, bRow, half);
  const int b = bh / HH, h = bh % HH;
  const int d = ct * 16 + r;
  if (rt < 13) {
#pragma unroll
    for (int i = 0; i < 8; ++i) {
      const int n = rt * 16 + i + 8 * half;
      const float val = (n < NN) ? acc[i] : 0.f;
      outb[((size_t)b * NP + n) * CC + h * DD + d] = f2bf(val);
    }
  }
}

// --------------- attention_output = out @ w_proj + b_proj ------------------
__global__ void __launch_bounds__(256) gemm_proj_kernel(
    const __bf16* __restrict__ outb, const __bf16* __restrict__ wpT,
    const float* __restrict__ bias, float* __restrict__ outp) {
  __shared__ __bf16 ldsA[16 * CC];                 // 24 KB A strip
  const int rt = blockIdx.x / 6;
  const int cg = blockIdx.x % 6;
  const int wave = threadIdx.x >> 5, lane = threadIdx.x & 31;
  const int half = lane >> 4, r = lane & 15;
  const int b = rt / 13, nt = rt % 13;
  if (wave == 0) {
    tdm_load_2d((uint32_t)(uintptr_t)ldsA,
                outb + (size_t)(b * NP + nt * 16) * CC, 16, CC);
    __builtin_amdgcn_s_wait_tensorcnt(0);
  }
  __syncthreads();
  const int colTile = cg * 8 + wave;               // 0..47
  const __bf16* aRow = ldsA + r * CC;
  const __bf16* bRow = wpT + (size_t)(colTile * 16 + r) * CC;
  v8f acc = wmma_tile<CC>(aRow, bRow, half);
  const int j = colTile * 16 + r;
  const float bj = bias[j];
#pragma unroll
  for (int i = 0; i < 8; ++i) {
    const int n = nt * 16 + i + 8 * half;
    if (n < NN) outp[((size_t)b * NN + n) * CC + j] = acc[i] + bj;
  }
}

// ---------------------------------------------------------------------------
extern "C" void kernel_launch(void* const* d_in, const int* in_sizes, int n_in,
                              void* d_out, int out_size, void* d_ws,
                              size_t ws_size, hipStream_t stream) {
  (void)in_sizes; (void)n_in; (void)out_size; (void)ws_size;
  const float* x     = (const float*)d_in[0];
  const float* wqkv  = (const float*)d_in[1];
  const float* wproj = (const float*)d_in[2];
  const float* bproj = (const float*)d_in[3];
  const float* wcl   = (const float*)d_in[4];
  const float* wcw   = (const float*)d_in[5];

  float* out_attn   = (float*)d_out;                         // [32,197,768]
  float* out_scores = out_attn + (size_t)BB * NN * CC;       // [32,12,197,197]
  float* out_probs  = out_scores + (size_t)BB * HH * NN * NN;
  float* out_vmap   = out_probs + (size_t)BB * HH * NN * NN;

  char* ws = (char*)d_ws;                                    // 16B-aligned subbuffers
  __bf16* xb    = (__bf16*)ws; ws += (size_t)BB * NP * CC * 2;
  __bf16* wqkvT = (__bf16*)ws; ws += (size_t)C3 * CC * 2;
  __bf16* wpT   = (__bf16*)ws; ws += (size_t)CC * CC * 2;
  __bf16* qb    = (__bf16*)ws; ws += (size_t)BH * NP * DD * 2;
  __bf16* kb    = (__bf16*)ws; ws += (size_t)BH * NP * DD * 2;
  __bf16* vb    = (__bf16*)ws; ws += (size_t)BH * NP * DD * 2;
  __bf16* vt    = (__bf16*)ws; ws += (size_t)BH * DD * NK * 2;
  __bf16* aw    = (__bf16*)ws; ws += (size_t)BH * NP * NK * 2;
  __bf16* outb  = (__bf16*)ws; ws += (size_t)BB * NP * CC * 2;
  // total ~103 MB of d_ws

  prep_x_kernel<<<(BB * NP * CC) / 256, 256, 0, stream>>>(x, xb);
  prep_wT_kernel<<<(C3 * CC) / 256, 256, 0, stream>>>(wqkv, wqkvT, CC, C3);
  prep_wT_kernel<<<(CC * CC) / 256, 256, 0, stream>>>(wproj, wpT, CC, CC);
  zero_u32_kernel<<<2048, 256, 0, stream>>>((uint32_t*)vt,
                                            (size_t)BH * DD * NK / 2);
  zero_u32_kernel<<<2048, 256, 0, stream>>>((uint32_t*)aw,
                                            (size_t)BH * NP * NK / 2);

  gemm_qkv_kernel<<<416 * 18, 256, 0, stream>>>(xb, wqkvT, qb, kb, vb, vt);
  gemm_nt_kernel<<<BH * 26, 256, 0, stream>>>(qb, kb, out_scores, SCALE);
  gemm_nt_kernel<<<BH * 26, 256, 0, stream>>>(vb, vb, out_vmap, SCALE);
  softmax_rows_kernel<<<(BH * NN) / 8, 256, 0, stream>>>(out_vmap, BH * NN);
  conv_softmax_kernel<<<BB * NN, 256, 0, stream>>>(out_scores, wcl, wcw,
                                                   out_probs, aw);
  gemm_av_kernel<<<BH * 7, 256, 0, stream>>>(aw, vt, outb);
  gemm_proj_kernel<<<416 * 6, 256, 0, stream>>>(outb, wpT, bproj, out_attn);
}
